// DoubleGINConv_87230785782146
// MI455X (gfx1250) — compile-verified, matrix operand order checked by reading, not run on hardware
//
#include <hip/hip_runtime.h>
#include <hip/hip_fp16.h>

typedef __attribute__((ext_vector_type(16))) _Float16 v16h;
typedef __attribute__((ext_vector_type(8)))  _Float16 v8h;
typedef __attribute__((ext_vector_type(8)))  float    v8f;

#define N_NODES 50000
#define N_EDGES 800000
#define IN_CH   96
#define OUT_CH  128

// ---------------- elementwise helpers ----------------

__global__ void k_f32_to_f16(const float* __restrict__ src,
                             _Float16* __restrict__ dst, int n) {
  int i = blockIdx.x * blockDim.x + threadIdx.x;
  if (i < n) dst[i] = (_Float16)src[i];
}

__global__ void k_zero_f32(float* __restrict__ dst, int n) {
  int i = blockIdx.x * blockDim.x + threadIdx.x;
  if (i < n) dst[i] = 0.0f;
}

// ---------------- edge aggregation: agg[dst] += x[src] ----------------
// One thread per (edge, 4 channels): b128 gather + 4 f32 atomics, all
// L2-resident on MI455X (node features + agg << 192MB L2).

template <int C>
__global__ void k_edge_scatter(const float* __restrict__ x,
                               const long long* __restrict__ eidx,
                               float* __restrict__ agg) {
  constexpr int C4 = C / 4;
  int tid = blockIdx.x * blockDim.x + threadIdx.x;
  const int total = N_EDGES * C4;
  if (tid >= total) return;
  int e = tid / C4;
  int c = (tid - e * C4) * 4;
  int s = (int)eidx[e];
  int d = (int)eidx[N_EDGES + e];
  const float4 v = *(const float4*)(x + (long long)s * C + c);
  float* p = agg + (long long)d * C + c;
  unsafeAtomicAdd(p + 0, v.x);
  unsafeAtomicAdd(p + 1, v.y);
  unsafeAtomicAdd(p + 2, v.z);
  unsafeAtomicAdd(p + 3, v.w);
}

// ---------------- WMMA GEMM ----------------
// out = relu( A[M,K] @ W[K,128] + b ),  A = (SUMF32 ? f32 Ax + f32 Ag : f16 Ah)
// Block = 256 threads = 8 waves; block owns 80 rows (5 M-tiles) x 128 cols.
// Per K-step: one B fragment from transposed-W LDS, then ALL five A fragments
// are loaded/converted (10 b128 loads in flight), then five back-to-back
// v_wmma (distinct D/A, shared B -> no WMMA data hazards between them).
// K is a template constant so the K-loop fully unrolls into immediate-offset
// addressing. M = 50000 = 625*80 exactly -> no tails, EXEC all-ones.

template <bool SUMF32, int K>
__launch_bounds__(256)
__global__ void k_gemm_bias_relu(const void* __restrict__ Ap0,   // f16 A  (or f32 x)
                                 const float* __restrict__ Ap1,  // f32 agg (SUMF32)
                                 const _Float16* __restrict__ W, // [K][128] f16
                                 const float* __restrict__ bias, // [128]
                                 _Float16* __restrict__ outH,
                                 float* __restrict__ outF) {
  constexpr int MT = 5;        // 16-row tiles per block
  constexpr int KP = K + 8;    // LDS row stride: 16B-aligned, bank-staggered
  __shared__ _Float16 sW[128 * KP];

  const int tid  = threadIdx.x;
  const int lane = tid & 31;
  const int wv   = tid >> 5;   // wave id -> N tile
  const int hs   = lane >> 4;  // half-wave select
  const int mn   = lane & 15;
  const int m0   = blockIdx.x * (MT * 16);
  const int col  = wv * 16 + mn;

  // One-time stage of W transposed into LDS.
  for (int idx = tid; idx < K * 128; idx += 256) {
    int k = idx >> 7;
    int n = idx & 127;
    sW[n * KP + k] = W[idx];
  }
  __syncthreads();

  // Hoisted per-tile fragment base pointers (row mn of each tile, +hs chunk).
  const float*    pxh[MT];
  const float*    pgh[MT];
  const _Float16* pah[MT];
#pragma unroll
  for (int t = 0; t < MT; ++t) {
    const long long rowoff = (long long)(m0 + t * 16 + mn) * K + hs * 8;
    if (SUMF32) {
      pxh[t] = (const float*)Ap0 + rowoff;
      pgh[t] = Ap1 + rowoff;
    } else {
      pah[t] = (const _Float16*)Ap0 + rowoff;
    }
  }
  const _Float16* pbh = sW + col * KP + hs * 8;

  v8f c[MT] = {};
#pragma unroll
  for (int kt = 0; kt < K / 32; ++kt) {
    // B fragment from LDS.
    v8h blo = *(const v8h*)(pbh + kt * 32);
    v8h bhi = *(const v8h*)(pbh + kt * 32 + 16);
    v16h b;
#pragma unroll
    for (int i = 0; i < 8; ++i) { b[i] = blo[i]; b[i + 8] = bhi[i]; }

    // Phase 1: all A fragments (loads overlap; converts fill latency).
    v16h a[MT];
#pragma unroll
    for (int t = 0; t < MT; ++t) {
      if (SUMF32) {
        v8f xlo = *(const v8f*)(pxh[t] + kt * 32);
        v8f xhi = *(const v8f*)(pxh[t] + kt * 32 + 16);
        v8f glo = *(const v8f*)(pgh[t] + kt * 32);
        v8f ghi = *(const v8f*)(pgh[t] + kt * 32 + 16);
#pragma unroll
        for (int i = 0; i < 8; ++i) {
          a[t][i]     = (_Float16)(xlo[i] + glo[i]);
          a[t][i + 8] = (_Float16)(xhi[i] + ghi[i]);
        }
      } else {
        v8h alo = *(const v8h*)(pah[t] + kt * 32);
        v8h ahi = *(const v8h*)(pah[t] + kt * 32 + 16);
#pragma unroll
        for (int i = 0; i < 8; ++i) { a[t][i] = alo[i]; a[t][i + 8] = ahi[i]; }
      }
    }

    // Phase 2: five hazard-free back-to-back WMMAs sharing B.
#pragma unroll
    for (int t = 0; t < MT; ++t) {
      c[t] = __builtin_amdgcn_wmma_f32_16x16x32_f16(
          false, a[t], false, b, (short)0, c[t], false, false);
    }
  }

  // Epilogue: vgpr r of lanes 0-15 -> row r, lanes 16-31 -> row 8+r.
  const float bv = bias[col];
#pragma unroll
  for (int t = 0; t < MT; ++t) {
    const int rowbase = m0 + t * 16 + hs * 8;
#pragma unroll
    for (int r = 0; r < 8; ++r) {
      float v = c[t][r] + bv;
      v = v > 0.0f ? v : 0.0f;
      long long o = (long long)(rowbase + r) * 128 + col;
      if (outF) outF[o] = v;
      if (outH) outH[o] = (_Float16)v;
    }
  }
}

// ---------------- driver ----------------

extern "C" void kernel_launch(void* const* d_in, const int* in_sizes, int n_in,
                              void* d_out, int out_size, void* d_ws, size_t ws_size,
                              hipStream_t stream) {
  const float*     x    = (const float*)d_in[0];
  const long long* eidx = (const long long*)d_in[1];
  const float* w1a = (const float*)d_in[2];
  const float* b1a = (const float*)d_in[3];
  const float* w1b = (const float*)d_in[4];
  const float* b1b = (const float*)d_in[5];
  const float* w2a = (const float*)d_in[6];
  const float* b2a = (const float*)d_in[7];
  const float* w2b = (const float*)d_in[8];
  const float* b2b = (const float*)d_in[9];
  float* out = (float*)d_out;

  // Workspace carve-up (~64 MB).
  char*  ws  = (char*)d_ws;
  size_t off = 0;
  auto carve = [&](size_t bytes) {
    void* p = ws + off;
    off += (bytes + 255) & ~(size_t)255;
    return p;
  };
  _Float16* hw1a = (_Float16*)carve((size_t)IN_CH  * OUT_CH * 2);
  _Float16* hw1b = (_Float16*)carve((size_t)OUT_CH * OUT_CH * 2);
  _Float16* hw2a = (_Float16*)carve((size_t)OUT_CH * OUT_CH * 2);
  _Float16* hw2b = (_Float16*)carve((size_t)OUT_CH * OUT_CH * 2);
  float*    aggb = (float*)   carve((size_t)N_NODES * OUT_CH * 4);
  _Float16* Tbuf = (_Float16*)carve((size_t)N_NODES * OUT_CH * 2);
  float*    h1   = (float*)   carve((size_t)N_NODES * OUT_CH * 4);

  const int B = 256;
  const int GEMM_GRID = N_NODES / 80;  // 625

  // f16 weight conversion (tiny)
  k_f32_to_f16<<<(IN_CH * OUT_CH + B - 1) / B, B, 0, stream>>>(w1a, hw1a, IN_CH * OUT_CH);
  k_f32_to_f16<<<(OUT_CH * OUT_CH + B - 1) / B, B, 0, stream>>>(w1b, hw1b, OUT_CH * OUT_CH);
  k_f32_to_f16<<<(OUT_CH * OUT_CH + B - 1) / B, B, 0, stream>>>(w2a, hw2a, OUT_CH * OUT_CH);
  k_f32_to_f16<<<(OUT_CH * OUT_CH + B - 1) / B, B, 0, stream>>>(w2b, hw2b, OUT_CH * OUT_CH);

  // ---- layer 1 ----
  const int n1 = N_NODES * IN_CH;   // 4.8M
  k_zero_f32<<<(n1 + B - 1) / B, B, 0, stream>>>(aggb, n1);
  k_edge_scatter<IN_CH>
      <<<(N_EDGES * (IN_CH / 4) + B - 1) / B, B, 0, stream>>>(x, eidx, aggb);
  k_gemm_bias_relu<true, IN_CH>
      <<<GEMM_GRID, B, 0, stream>>>(x, aggb, hw1a, b1a, Tbuf, nullptr);
  k_gemm_bias_relu<false, OUT_CH>
      <<<GEMM_GRID, B, 0, stream>>>(Tbuf, nullptr, hw1b, b1b, nullptr, h1);

  // ---- layer 2 ----
  const int n2 = N_NODES * OUT_CH;  // 6.4M
  k_zero_f32<<<(n2 + B - 1) / B, B, 0, stream>>>(aggb, n2);
  k_edge_scatter<OUT_CH>
      <<<(N_EDGES * (OUT_CH / 4) + B - 1) / B, B, 0, stream>>>(h1, eidx, aggb);
  k_gemm_bias_relu<true, OUT_CH>
      <<<GEMM_GRID, B, 0, stream>>>(h1, aggb, hw2a, b2a, Tbuf, nullptr);
  k_gemm_bias_relu<false, OUT_CH>
      <<<GEMM_GRID, B, 0, stream>>>(Tbuf, nullptr, hw2b, b2b, nullptr, out);
}
